// __Deprecated_90546500534756
// MI455X (gfx1250) — matrix operand order repaired and checked by reading, not compiled, not run on hardware
//
#include <hip/hip_runtime.h>

typedef _Float16 h16;
typedef __attribute__((ext_vector_type(8)))  _Float16 v8h;
typedef __attribute__((ext_vector_type(16))) _Float16 v16h;
typedef __attribute__((ext_vector_type(8)))  float    v8f;

#define WMMA16(a, b, c) \
  __builtin_amdgcn_wmma_f32_16x16x32_f16(false, (a), false, (b), (short)0, (c), false, false)

// ---------------- LDS layout (bytes), one workgroup = one batch ----------------
#define ASUM_OFF    0          // h16[128*136]  row-major Asum
#define INVDEG_OFF  34816      // float[128]
#define STAGE_OFF   35328      // float[8*640] per-wave edge staging (P0 only)
#define X1_OFF      35328      // h16[128*40]   (P1-P2, reuses stage)
#define H1A_OFF     35328      // h16[128*40]   (P3-P4, reuses X1)
#define H1CM_OFF    45568      // h16[32*136]   (P2-P3)
#define RED_OFF     45568      // float[256]    (P7, reuses h1cm)
#define H2CM_OFF    55808      // h16[64*136]
#define H2ACM_OFF   73216      // h16[64*136]
#define N128_OFF    90624      // h16[128*136]  [h2a | ev] row-major
#define SMEM_BYTES  125440

// ---------------- workspace layout (bytes) ----------------
#define WS_M1T    0      // h16[32*32]   (X1(13,pad32) -> 32), stored [n][k]
#define WS_W2T    2048   // h16[64*32]   w_g2 transposed [n][k]
#define WS_C1     6144   // float[32]    fused g1 bias
#define WS_W128G  6272   // float[128]   folded g3a*g3b*head vector
#define WS_V10    6784   // float[16]    folded cond-path vector
#define WS_S      6848   // float[1]     folded scalar bias

// A-operand: element [m][k], storage K-contiguous with leading dim ld (halves).
__device__ __forceinline__ v16h ld_opA(const h16* p, int ld, int m0, int k0, int lane) {
  const h16* q = p + (size_t)(m0 + (lane & 15)) * ld + k0 + ((lane >> 4) << 3);
  v8h lo = *(const v8h*)q;
  v8h hi = *(const v8h*)(q + 16);
  return __builtin_shufflevector(lo, hi, 0,1,2,3,4,5,6,7,8,9,10,11,12,13,14,15);
}
// B-operand: element [k][n] fetched from storage [n][k] (K-contiguous), ld in halves.
__device__ __forceinline__ v16h ld_opB(const h16* p, int ld, int n0, int k0, int lane) {
  const h16* q = p + (size_t)(n0 + (lane & 15)) * ld + k0 + ((lane >> 4) << 3);
  v8h lo = *(const v8h*)q;
  v8h hi = *(const v8h*)(q + 16);
  return __builtin_shufflevector(lo, hi, 0,1,2,3,4,5,6,7,8,9,10,11,12,13,14,15);
}

// =================== prep: fold all weights once per launch ===================
__global__ __launch_bounds__(256) void prep_kernel(
    const float* __restrict__ w_ne0, const float* __restrict__ b_ne0,
    const float* __restrict__ w_ne1, const float* __restrict__ b_ne1,
    const float* __restrict__ w_g1,  const float* __restrict__ b_g1,
    const float* __restrict__ w_g2,
    const float* __restrict__ w_g3a, const float* __restrict__ b_g3a,
    const float* __restrict__ w_g3b, const float* __restrict__ b_g3b,
    const float* __restrict__ w_ce0, const float* __restrict__ b_ce0,
    const float* __restrict__ w_ce1, const float* __restrict__ b_ce1,
    const float* __restrict__ w_f0,  const float* __restrict__ b_f0,
    const float* __restrict__ w_f1,  const float* __restrict__ b_f1,
    const float* __restrict__ w_f2,  const float* __restrict__ b_f2,
    const float* __restrict__ w_f3,  const float* __restrict__ b_f3,
    h16* __restrict__ M1T, h16* __restrict__ W2T,
    float* __restrict__ c1, float* __restrict__ w128g,
    float* __restrict__ v10, float* __restrict__ Sout) {
  __shared__ float Wenc[8 * 32], benc[32], Wf23[64], Wf123[128], u[128], vt[16], w256[256];
  const int t = threadIdx.x;

  { // Wenc = w_ne0(8x64) @ w_ne1(64x32); benc = b_ne0@w_ne1 + b_ne1
    int p = t >> 5, n = t & 31;
    float s = 0.f;
    for (int m = 0; m < 64; ++m) s += w_ne0[p * 64 + m] * w_ne1[m * 32 + n];
    Wenc[p * 32 + n] = s;
  }
  if (t < 32) {
    float s = 0.f;
    for (int m = 0; m < 64; ++m) s += b_ne0[m] * w_ne1[m * 32 + t];
    benc[t] = s + b_ne1[t];
  }
  __syncthreads();

  // M1 (32pad x 32): rows 0..4 = w_g1[0:5], rows 5..12 = Wenc@w_g1[5:37], rest 0. Stored [n][k].
  for (int idx = t; idx < 32 * 32; idx += 256) {
    int n = idx >> 5, k = idx & 31;
    float v = 0.f;
    if (k < 5) v = w_g1[k * 32 + n];
    else if (k < 13) {
      for (int m = 0; m < 32; ++m) v += Wenc[(k - 5) * 32 + m] * w_g1[(5 + m) * 32 + n];
    }
    M1T[n * 32 + k] = (h16)v;
  }
  if (t < 32) {
    float s = b_g1[t];
    for (int m = 0; m < 32; ++m) s += benc[m] * w_g1[(5 + m) * 32 + t];
    c1[t] = s;
  }
  for (int idx = t; idx < 64 * 32; idx += 256) {  // W2T[n][k] = w_g2[k][n]
    int n = idx >> 5, k = idx & 31;
    W2T[n * 32 + k] = (h16)w_g2[k * 64 + n];
  }

  // fold the head:  out = <gl128,u> + <cond,v10> + S
  if (t < 64) { float s = 0.f; for (int j = 0; j < 32; ++j) s += w_f2[t * 32 + j] * w_f3[j]; Wf23[t] = s; }
  __syncthreads();
  if (t < 128) { float s = 0.f; for (int j = 0; j < 64; ++j) s += w_f1[t * 64 + j] * Wf23[j]; Wf123[t] = s; }
  __syncthreads();
  if (t < 128) { float s = 0.f; for (int j = 0; j < 128; ++j) s += w_f0[t * 128 + j] * Wf123[j]; u[t] = s; }
  if (t >= 128 && t < 144) {
    int r = t - 128; float s = 0.f;
    for (int j = 0; j < 128; ++j) s += w_f0[(128 + r) * 128 + j] * Wf123[j];
    vt[r] = s;
  }
  __syncthreads();
  { float s = 0.f; for (int j = 0; j < 128; ++j) s += w_g3b[t * 128 + j] * u[j]; w256[t] = s; }
  __syncthreads();
  if (t < 128) { float s = 0.f; for (int j = 0; j < 256; ++j) s += w_g3a[t * 256 + j] * w256[j]; w128g[t] = s; }
  if (t < 16) {
    float s = 0.f;
    if (t < 10) {
      for (int tt = 0; tt < 16; ++tt) {
        float wc = 0.f;
        for (int m = 0; m < 32; ++m) wc += w_ce0[t * 32 + m] * w_ce1[m * 16 + tt];
        s += wc * vt[tt];
      }
    }
    v10[t] = s;
  }
  if (t == 0) {
    float S = b_f3[0];
    for (int j = 0; j < 32; ++j)  S += b_f2[j] * w_f3[j];
    for (int j = 0; j < 64; ++j)  S += b_f1[j] * Wf23[j];
    for (int j = 0; j < 128; ++j) S += b_f0[j] * Wf123[j];
    for (int tt = 0; tt < 16; ++tt) {
      float bct = b_ce1[tt];
      for (int m = 0; m < 32; ++m) bct += b_ce0[m] * w_ce1[m * 16 + tt];
      S += bct * vt[tt];
    }
    for (int j = 0; j < 128; ++j) S += b_g3b[j] * u[j];
    for (int j = 0; j < 256; ++j) S += b_g3a[j] * w256[j];
    *Sout = S;
  }
}

// =================== main: one workgroup (8 waves) per batch ===================
__global__ __launch_bounds__(256) void gnn_batch_kernel(
    const float* __restrict__ edges, const float* __restrict__ nodes,
    const float* __restrict__ nparams, const float* __restrict__ cond,
    const float* __restrict__ b_g2,
    const h16* __restrict__ M1T, const h16* __restrict__ W2T,
    const float* __restrict__ c1, const float* __restrict__ w128g,
    const float* __restrict__ v10, const float* __restrict__ Sconst,
    float* __restrict__ out) {
  extern __shared__ char smem[];
  h16*   asum   = (h16*)(smem + ASUM_OFF);     // [128][136]
  float* invdeg = (float*)(smem + INVDEG_OFF); // [128]
  float* stage  = (float*)(smem + STAGE_OFF);  // [8][640]
  h16*   X1     = (h16*)(smem + X1_OFF);       // [128][40]
  h16*   h1a    = (h16*)(smem + H1A_OFF);      // [128][40]
  h16*   h1cm   = (h16*)(smem + H1CM_OFF);     // [32][136]
  float* red    = (float*)(smem + RED_OFF);    // [256]
  h16*   h2cm   = (h16*)(smem + H2CM_OFF);     // [64][136]
  h16*   h2acm  = (h16*)(smem + H2ACM_OFF);    // [64][136]
  h16*   n128   = (h16*)(smem + N128_OFF);     // [128][136]

  const int b = blockIdx.x;
  const int t = threadIdx.x;
  const int w = t >> 5, l = t & 31;

  // ---- P0: stream edges -> Asum(f16,row-major) + invdeg. One wave per row. ----
  {
    float*  mystage = stage + w * 640;
    float4* st4     = (float4*)mystage;
    for (int i = w; i < 128; i += 8) {
      const float4* src = (const float4*)(edges + (size_t)(b * 128 + i) * 640);
      if (i + 8 < 128)
        __builtin_prefetch(edges + (size_t)(b * 128 + i + 8) * 640 + l * 20, 0, 0);
#pragma unroll
      for (int k = 0; k < 5; ++k) st4[k * 32 + l] = src[k * 32 + l];
      asm volatile("s_wait_dscnt 0" ::: "memory");
      float rs = 0.f;
#pragma unroll
      for (int k = 0; k < 4; ++k) {
        int j = l + 32 * k;
        const float* e = mystage + j * 5;
        float s = e[1] + e[2] + e[3] + e[4];  // edge types 1..4 (type 0 dropped)
        asum[i * 136 + j] = (h16)s;
        rs += s;
      }
      for (int off = 16; off > 0; off >>= 1) rs += __shfl_down(rs, off);
      if (l == 0) invdeg[i] = 1.f / fmaxf(rs, 1e-8f);
    }
  }
  __syncthreads();

  // ---- P1: build X1 = [nodes(5) | node_params(8) | zeros] row-major [128][40] ----
  for (int idx = t; idx < 128 * 40; idx += 256) {
    int i = idx / 40, c = idx - i * 40;
    float v = 0.f;
    if (c < 5) v = nodes[(size_t)(b * 128 + i) * 5 + c];
    else if (c < 13) v = nparams[(size_t)(b * 128 + i) * 8 + (c - 5)];
    X1[idx] = (h16)v;
  }
  __syncthreads();

  // ---- P2: h1 = X1 @ M1 + c1   (128x32x32) -> h1 col-major [feat32][node] ----
  {
    int m0 = w * 16;
    v16h a = ld_opA(X1, 40, m0, 0, l);
#pragma unroll
    for (int nt = 0; nt < 2; ++nt) {
      v8f c = {};
      v16h bb = ld_opB(M1T, 32, nt * 16, 0, l);
      c = WMMA16(a, bb, c);
      int col = nt * 16 + (l & 15);
      int r0  = m0 + ((l >> 4) << 3);
      float bias = c1[col];
      v8h o;
#pragma unroll
      for (int r = 0; r < 8; ++r) o[r] = (h16)(c[r] + bias);
      *(v8h*)(h1cm + col * 136 + r0) = o;
    }
  }
  __syncthreads();

  // ---- P3: h1a = (Asum@h1)/deg, transposed: h1^T · Asum^T -> row-major [node][32] ----
  {
    int n0 = w * 16;  // node tile
    v16h bAs[4];
#pragma unroll
    for (int kk = 0; kk < 4; ++kk) bAs[kk] = ld_opB(asum, 136, n0, kk * 32, l);
#pragma unroll
    for (int mt = 0; mt < 2; ++mt) {
      v8f c = {};
#pragma unroll
      for (int kk = 0; kk < 4; ++kk) {
        v16h a = ld_opA(h1cm, 136, mt * 16, kk * 32, l);
        c = WMMA16(a, bAs[kk], c);
      }
      int i  = n0 + (l & 15);
      int d0 = mt * 16 + ((l >> 4) << 3);
      float sc = invdeg[i];
      v8h o;
#pragma unroll
      for (int r = 0; r < 8; ++r) o[r] = (h16)(c[r] * sc);
      *(v8h*)(h1a + i * 40 + d0) = o;
    }
  }
  __syncthreads();

  // ---- P4: h2 = h1a @ W2 + b2  (128x32x64) -> h2 col-major [feat64][node] ----
  {
    int m0 = w * 16;
    v16h a = ld_opA(h1a, 40, m0, 0, l);
#pragma unroll
    for (int nt = 0; nt < 4; ++nt) {
      v8f c = {};
      v16h bb = ld_opB(W2T, 32, nt * 16, 0, l);
      c = WMMA16(a, bb, c);
      int col = nt * 16 + (l & 15);
      int r0  = m0 + ((l >> 4) << 3);
      float bias = b_g2[col];
      v8h o;
#pragma unroll
      for (int r = 0; r < 8; ++r) o[r] = (h16)(c[r] + bias);
      *(v8h*)(h2cm + col * 136 + r0) = o;
    }
  }
  __syncthreads();

  // ---- P5: h2a = (Asum@h2)/deg -> n128[:, :64] (row-major) + col-major copy ----
  {
    int n0 = w * 16;
    v16h bAs[4];
#pragma unroll
    for (int kk = 0; kk < 4; ++kk) bAs[kk] = ld_opB(asum, 136, n0, kk * 32, l);
#pragma unroll
    for (int mt = 0; mt < 4; ++mt) {
      v8f c = {};
#pragma unroll
      for (int kk = 0; kk < 4; ++kk) {
        v16h a = ld_opA(h2cm, 136, mt * 16, kk * 32, l);
        c = WMMA16(a, bAs[kk], c);
      }
      int i  = n0 + (l & 15);
      int d0 = mt * 16 + ((l >> 4) << 3);
      float sc = invdeg[i];
      v8h o;
#pragma unroll
      for (int r = 0; r < 8; ++r) o[r] = (h16)(c[r] * sc);
      *(v8h*)(n128 + i * 136 + d0) = o;  // row-major h2a
#pragma unroll
      for (int r = 0; r < 8; ++r) h2acm[(d0 + r) * 136 + i] = o[r];  // col-major copy
    }
  }
  __syncthreads();

  // ---- P6: h2aa = (Asum@h2a)/deg ; ev = h2a - h2aa -> n128[:, 64:128] ----
  {
    int n0 = w * 16;
    v16h bAs[4];
#pragma unroll
    for (int kk = 0; kk < 4; ++kk) bAs[kk] = ld_opB(asum, 136, n0, kk * 32, l);
#pragma unroll
    for (int mt = 0; mt < 4; ++mt) {
      v8f c = {};
#pragma unroll
      for (int kk = 0; kk < 4; ++kk) {
        v16h a = ld_opA(h2acm, 136, mt * 16, kk * 32, l);
        c = WMMA16(a, bAs[kk], c);
      }
      int i  = n0 + (l & 15);
      int d0 = mt * 16 + ((l >> 4) << 3);
      float sc = invdeg[i];
      v8h h2av = *(const v8h*)(n128 + i * 136 + d0);
      v8h o;
#pragma unroll
      for (int r = 0; r < 8; ++r) o[r] = (h16)((float)h2av[r] - c[r] * sc);
      *(v8h*)(n128 + i * 136 + 64 + d0) = o;
    }
  }
  __syncthreads();

  // ---- P7: out[b] = (1/128)*sum_i <n128[i], w128g> + <cond[b], v10> + S ----
  {
    int i = t >> 1, c0 = (t & 1) * 64;
    const h16* row = n128 + i * 136 + c0;
    float p = 0.f;
#pragma unroll 8
    for (int c = 0; c < 64; ++c) p += (float)row[c] * w128g[c0 + c];
    red[t] = p;
    __syncthreads();
    for (int s = 128; s > 0; s >>= 1) {
      if (t < s) red[t] += red[t + s];
      __syncthreads();
    }
    if (t == 0) {
      float v = red[0] * (1.f / 128.f) + *Sconst;
      for (int k = 0; k < 10; ++k) v += cond[(size_t)b * 10 + k] * v10[k];
      out[b] = v;
    }
  }
}

extern "C" void kernel_launch(void* const* d_in, const int* in_sizes, int n_in,
                              void* d_out, int out_size, void* d_ws, size_t ws_size,
                              hipStream_t stream) {
  const float* edges   = (const float*)d_in[0];
  const float* nodes   = (const float*)d_in[1];
  const float* nparams = (const float*)d_in[2];
  const float* cond    = (const float*)d_in[3];
  const float* w_ne0 = (const float*)d_in[4];   const float* b_ne0 = (const float*)d_in[5];
  const float* w_ne1 = (const float*)d_in[6];   const float* b_ne1 = (const float*)d_in[7];
  const float* w_g1  = (const float*)d_in[8];   const float* b_g1  = (const float*)d_in[9];
  const float* w_g2  = (const float*)d_in[10];  const float* b_g2  = (const float*)d_in[11];
  const float* w_g3a = (const float*)d_in[12];  const float* b_g3a = (const float*)d_in[13];
  const float* w_g3b = (const float*)d_in[14];  const float* b_g3b = (const float*)d_in[15];
  const float* w_ce0 = (const float*)d_in[16];  const float* b_ce0 = (const float*)d_in[17];
  const float* w_ce1 = (const float*)d_in[18];  const float* b_ce1 = (const float*)d_in[19];
  const float* w_f0  = (const float*)d_in[20];  const float* b_f0  = (const float*)d_in[21];
  const float* w_f1  = (const float*)d_in[22];  const float* b_f1  = (const float*)d_in[23];
  const float* w_f2  = (const float*)d_in[24];  const float* b_f2  = (const float*)d_in[25];
  const float* w_f3  = (const float*)d_in[26];  const float* b_f3  = (const float*)d_in[27];

  char* ws = (char*)d_ws;
  h16*   M1T   = (h16*)(ws + WS_M1T);
  h16*   W2T   = (h16*)(ws + WS_W2T);
  float* c1    = (float*)(ws + WS_C1);
  float* w128g = (float*)(ws + WS_W128G);
  float* v10   = (float*)(ws + WS_V10);
  float* S     = (float*)(ws + WS_S);

  const int B = in_sizes[3] / 10;  // cond is [B,10]

  prep_kernel<<<1, 256, 0, stream>>>(
      w_ne0, b_ne0, w_ne1, b_ne1, w_g1, b_g1, w_g2,
      w_g3a, b_g3a, w_g3b, b_g3b, w_ce0, b_ce0, w_ce1, b_ce1,
      w_f0, b_f0, w_f1, b_f1, w_f2, b_f2, w_f3, b_f3,
      M1T, W2T, c1, w128g, v10, S);

  gnn_batch_kernel<<<B, 256, SMEM_BYTES, stream>>>(
      edges, nodes, nparams, cond, b_g2,
      M1T, W2T, c1, w128g, v10, S, (float*)d_out);
}
